// ContrastiveLoss_47725676593592
// MI455X (gfx1250) — matrix operand order repaired and checked by reading, not compiled, not run on hardware
//
#include <hip/hip_runtime.h>

// ---------------------------------------------------------------------------
// SupCon contrastive loss, fused for MI455X (gfx1250, wave32, WMMA).
//   F = normalize([anchor; positive])  (bf16, 8192x1024)
//   sim = F F^T / T as 128x128 bf16-WMMA tiles (lower triangle only; sim is
//   symmetric), exp() + diagonal mask + row/col reduction fused in-register.
//   K-stages are double-buffered in LDS and filled with CDNA5
//   global_load_async_to_lds_b128 (ASYNCcnt), overlapping global traffic with
//   WMMA. Deterministic: each block writes exactly one partial slot per row.
// ---------------------------------------------------------------------------

typedef __attribute__((ext_vector_type(16))) __bf16         v16bf;
typedef __attribute__((ext_vector_type(8)))  float          v8f;
typedef __attribute__((ext_vector_type(4)))  unsigned int   u32x4;
typedef __attribute__((ext_vector_type(4)))  float          f32x4;
typedef __attribute__((ext_vector_type(4)))  unsigned short u16x4;

#define TEMP_INV 14.285714285714285714f  // 1 / 0.07
#define LDK 72                            // LDS row stride in halves (64 + 8 pad)

__device__ __forceinline__ unsigned short f2bf(float f) {
  unsigned int u = __float_as_uint(f);
  u += 0x7fffu + ((u >> 16) & 1u);        // round-to-nearest-even
  return (unsigned short)(u >> 16);
}
__device__ __forceinline__ float bf2f(unsigned short h) {
  return __uint_as_float(((unsigned int)h) << 16);
}

// CDNA5 async global->LDS copy, 16B per lane, ASYNCcnt-tracked (ISA 10.x).
// lds_addr: LDS byte offset (low 32 bits of generic pointer); GVS mode:
// mem = SADDR(base) + VADDR(goff).
__device__ __forceinline__ void async_ld16(unsigned lds_addr,
                                           const unsigned short* base,
                                           unsigned goff) {
  asm volatile("global_load_async_to_lds_b128 %0, %1, %2"
               :: "v"(lds_addr), "v"(goff), "s"(base)
               : "memory");
}

// --- kernel 1: L2-normalize each row of [anchor; positive] -> bf16 feats ----
__global__ __launch_bounds__(256)
void normalize_bf16_kernel(const float* __restrict__ anchor,
                           const float* __restrict__ pos,
                           unsigned short* __restrict__ feats,
                           int B, int D) {
  const int row = blockIdx.x;  // 0 .. 2B-1
  const float* __restrict__ src =
      (row < B) ? (anchor + (size_t)row * D) : (pos + (size_t)(row - B) * D);
  const int t = threadIdx.x;
  const f32x4* s4 = (const f32x4*)src;
  const int nvec = D >> 2;
  float ss = 0.f;
  for (int i = t; i < nvec; i += 256) {
    f32x4 x = s4[i];
    ss += x.x * x.x + x.y * x.y + x.z * x.z + x.w * x.w;
  }
#pragma unroll
  for (int m = 16; m >= 1; m >>= 1) ss += __shfl_xor(ss, m);
  __shared__ float wsum[8];
  if ((t & 31) == 0) wsum[t >> 5] = ss;
  __syncthreads();
  float tot = 0.f;
#pragma unroll
  for (int i = 0; i < 8; i++) tot += wsum[i];
  const float rn = rsqrtf(tot);
  u16x4* dst = (u16x4*)(feats + (size_t)row * D);
  for (int i = t; i < nvec; i += 256) {
    f32x4 x = s4[i];
    u16x4 o;
    o.x = f2bf(x.x * rn); o.y = f2bf(x.y * rn);
    o.z = f2bf(x.z * rn); o.w = f2bf(x.w * rn);
    dst[i] = o;
  }
}

// --- kernel 2: cross[i] = <a_i, p_i> / T -----------------------------------
__global__ __launch_bounds__(256)
void cross_kernel(const unsigned short* __restrict__ feats,
                  float* __restrict__ cross, int B, int D) {
  const int i = blockIdx.x;
  const int t = threadIdx.x;
  const u16x4* pa = (const u16x4*)(feats + (size_t)i * D);
  const u16x4* pp = (const u16x4*)(feats + (size_t)(i + B) * D);
  const int nvec = D >> 2;
  float d = 0.f;
  for (int k = t; k < nvec; k += 256) {
    u16x4 a = pa[k], p = pp[k];
    d += bf2f(a.x) * bf2f(p.x) + bf2f(a.y) * bf2f(p.y) +
         bf2f(a.z) * bf2f(p.z) + bf2f(a.w) * bf2f(p.w);
  }
#pragma unroll
  for (int m = 16; m >= 1; m >>= 1) d += __shfl_xor(d, m);
  __shared__ float wsum[8];
  if ((t & 31) == 0) wsum[t >> 5] = d;
  __syncthreads();
  if (t == 0) {
    float tot = 0.f;
#pragma unroll
    for (int j = 0; j < 8; j++) tot += wsum[j];
    cross[i] = tot * TEMP_INV;
  }
}

// --- kernel 3: WMMA tile GEMM + fused exp/row-sum (lower triangle) ---------
// 128x128 block tile, 128 threads = 4 waves, each wave owns a 64x64 tile
// (4x4 16x16 accumulators -> 1.0 ds_load_b128 per WMMA, high operand reuse).
union FragU { v16bf v; u32x4 q[2]; };

__global__ __launch_bounds__(128)
void simexp_kernel(const unsigned short* __restrict__ F,
                   float* __restrict__ partials,
                   int D, int nblk) {
  const int bn = blockIdx.x;
  const int bm = blockIdx.y;
  if (bn > bm) return;                 // sim is symmetric: lower triangle only

  __shared__ unsigned short lA[2][128 * LDK];   // double-buffered K stages
  __shared__ unsigned short lB[2][128 * LDK];
  __shared__ float rp[2][128];         // per-waveN row partials
  __shared__ float cp[2][128];         // per-waveM col partials (mirror)

  const int tid   = threadIdx.x;
  const int lane  = tid & 31;
  const int wid   = tid >> 5;          // 0..3
  const int waveM = wid >> 1;          // 0..1 -> 64-row strip
  const int waveN = wid & 1;           // 0..1 -> 64-col strip
  const int nlo   = lane & 15;

  v8f acc[4][4] = {};

  const int rowA0 = bm * 128;
  const int rowB0 = bn * 128;

  // Async staging: 128 rows x 64 halves per matrix = 1024 16B chunks each.
  // Thread t stages chunks j*128+t, j=0..7 for A and j=0..7 for B.
  // Per wave: 16 async issues per stage -> s_wait_asynccnt 16 == prev stage done.
  const int KB = D >> 6;               // number of 64-wide K stages

  auto issue_stage = [&](int buf, int k0) {
#pragma unroll
    for (int j = 0; j < 8; j++) {
      const int c   = j * 128 + tid;   // chunk id 0..1023
      const int row = c >> 2;
      const int cc  = (c & 3) * 8;     // half offset within 64-wide stage
      const unsigned la =
          (unsigned)(uintptr_t)&lA[buf][row * LDK + cc];
      async_ld16(la, F, (unsigned)((rowA0 + row) * D + k0 + cc) * 2u);
      const unsigned lb =
          (unsigned)(uintptr_t)&lB[buf][row * LDK + cc];
      async_ld16(lb, F, (unsigned)((rowB0 + row) * D + k0 + cc) * 2u);
    }
  };

  auto compute_stage = [&](int buf) {
#pragma unroll
    for (int ks = 0; ks < 2; ks++) {
      // B fragment: lane n holds 16 contiguous K-halves at (lane<16 ? 0 : 16)
      FragU bfr[4];
      const int kbB = ks * 32 + ((lane < 16) ? 0 : 16);
#pragma unroll
      for (int nt = 0; nt < 4; nt++) {
        const unsigned short* p =
            &lB[buf][(waveN * 64 + nt * 16 + nlo) * LDK + kbB];
        bfr[nt].q[0] = *(const u32x4*)p;
        bfr[nt].q[1] = *(const u32x4*)(p + 8);
      }
      // A fragment: lane m holds 16 contiguous K-halves at (lane<16 ? 0 : 16)
      const int kbA = ks * 32 + ((lane < 16) ? 0 : 16);
#pragma unroll
      for (int mt = 0; mt < 4; mt++) {
        const unsigned short* p =
            &lA[buf][(waveM * 64 + mt * 16 + nlo) * LDK + kbA];
        FragU af;
        af.q[0] = *(const u32x4*)p;
        af.q[1] = *(const u32x4*)(p + 8);
#pragma unroll
        for (int nt = 0; nt < 4; nt++) {
          acc[mt][nt] = __builtin_amdgcn_wmma_f32_16x16x32_bf16(
              false, af.v, false, bfr[nt].v, (short)0, acc[mt][nt],
              false, false);
        }
      }
    }
  };

  issue_stage(0, 0);
  int kb = 0;
  for (; kb < KB - 1; kb++) {
    issue_stage((kb + 1) & 1, (kb + 1) * 64);
    asm volatile("s_wait_asynccnt 0x10" ::: "memory");  // prev stage landed
    __syncthreads();
    compute_stage(kb & 1);
    __syncthreads();                                    // readers done
  }
  asm volatile("s_wait_asynccnt 0x0" ::: "memory");
  __syncthreads();
  compute_stage(kb & 1);

  // Epilogue: e = exp(sim/T) with diag masked; reduce rows (and cols for the
  // mirrored upper-triangle contribution). C/D layout: row = v + 8*(lane>=16),
  // col = lane%16.
  const int g8 = (lane >> 4) << 3;
  float csum[4] = {0.f, 0.f, 0.f, 0.f};
#pragma unroll
  for (int mt = 0; mt < 4; mt++) {
#pragma unroll
    for (int v = 0; v < 8; v++) {
      const int rloc = waveM * 64 + mt * 16 + v + g8;
      const int gr   = rowA0 + rloc;
      float rv = 0.f;
#pragma unroll
      for (int nt = 0; nt < 4; nt++) {
        const int gc = rowB0 + waveN * 64 + nt * 16 + nlo;
        float e = (gr == gc) ? 0.f : __expf(acc[mt][nt][v] * TEMP_INV);
        csum[nt] += e;
        rv += e;
      }
      rv += __shfl_xor(rv, 1);          // reduce 16 cols across half-wave
      rv += __shfl_xor(rv, 2);
      rv += __shfl_xor(rv, 4);
      rv += __shfl_xor(rv, 8);
      if (nlo == 0) rp[waveN][rloc] = rv;
    }
  }
#pragma unroll
  for (int nt = 0; nt < 4; nt++) {
    csum[nt] += __shfl_xor(csum[nt], 16);  // combine the two 8-row groups
    if (lane < 16) cp[waveM][waveN * 64 + nt * 16 + lane] = csum[nt];
  }
  __syncthreads();
  {
    float s = rp[0][tid] + rp[1][tid];
    partials[(size_t)(rowA0 + tid) * nblk + bn] = s;   // row contribution
    if (bm != bn) {                                    // mirrored contribution
      float c = cp[0][tid] + cp[1][tid];
      partials[(size_t)(rowB0 + tid) * nblk + bm] = c;
    }
  }
}

// --- kernel 4: rowsum[r] = sum_j partials[r][j] ----------------------------
__global__ void rowsum_kernel(const float* __restrict__ partials,
                              float* __restrict__ rowsum, int rows, int nblk) {
  int r = blockIdx.x * blockDim.x + threadIdx.x;
  if (r < rows) {
    const float* p = partials + (size_t)r * nblk;
    float s = 0.f;
    for (int j = 0; j < nblk; j++) s += p[j];
    rowsum[r] = s;
  }
}

// --- kernel 5: final scalar loss -------------------------------------------
__global__ __launch_bounds__(256)
void loss_kernel(const float* __restrict__ rowsum,
                 const float* __restrict__ cross,
                 const int* __restrict__ labels,
                 float* __restrict__ out, int B) {
  const int t = threadIdx.x;
  float tot = 0.f, np = 0.f;
  for (int r = t; r < 2 * B; r += 256) {
    const int i = (r < B) ? r : (r - B);
    const float lab = (float)labels[i];
    tot += (__logf(rowsum[r]) - cross[i]) * lab;   // lse - pos_sim
  }
  for (int i = t; i < B; i += 256) np += (float)labels[i];
#pragma unroll
  for (int m = 16; m >= 1; m >>= 1) {
    tot += __shfl_xor(tot, m);
    np  += __shfl_xor(np, m);
  }
  __shared__ float st[8], sn[8];
  if ((t & 31) == 0) { st[t >> 5] = tot; sn[t >> 5] = np; }
  __syncthreads();
  if (t == 0) {
    float T = 0.f, N = 0.f;
#pragma unroll
    for (int i = 0; i < 8; i++) { T += st[i]; N += sn[i]; }
    N *= 2.f;                                      // labels counted twice
    out[0] = (N > 0.f) ? (T / N) : 0.f;
  }
}

// ---------------------------------------------------------------------------
extern "C" void kernel_launch(void* const* d_in, const int* in_sizes, int n_in,
                              void* d_out, int out_size, void* d_ws,
                              size_t ws_size, hipStream_t stream) {
  const float* anchor = (const float*)d_in[0];
  const float* pos    = (const float*)d_in[1];
  const int*   labels = (const int*)d_in[2];
  float* out = (float*)d_out;

  const int B    = in_sizes[2];      // 4096
  const int D    = in_sizes[0] / B;  // 1024
  const int rows = 2 * B;            // 8192
  const int nblk = rows / 128;       // 64 tile-strips

  char* ws = (char*)d_ws;
  size_t off = 0;
  auto take = [&](size_t bytes) -> void* {
    void* p = ws + off;
    off = (off + bytes + 255) & ~(size_t)255;
    return p;
  };
  unsigned short* feats    = (unsigned short*)take((size_t)rows * D * 2); // 16 MiB
  float*          cross    = (float*)take((size_t)B * 4);                 // 16 KiB
  float*          partials = (float*)take((size_t)rows * nblk * 4);       //  2 MiB
  float*          rowsum   = (float*)take((size_t)rows * 4);              // 32 KiB
  (void)ws_size; (void)n_in; (void)out_size;

  normalize_bf16_kernel<<<rows, 256, 0, stream>>>(anchor, pos, feats, B, D);
  cross_kernel<<<B, 256, 0, stream>>>(feats, cross, B, D);
  simexp_kernel<<<dim3(nblk, nblk), 128, 0, stream>>>(feats, partials, D, nblk);
  rowsum_kernel<<<(rows + 255) / 256, 256, 0, stream>>>(partials, rowsum, rows, nblk);
  loss_kernel<<<1, 256, 0, stream>>>(rowsum, cross, labels, out, B);
}
